// HeteroRGNN_64725157151126
// MI455X (gfx1250) — matrix operand order repaired and checked by reading, not compiled, not run on hardware
//
#include <hip/hip_runtime.h>
#include <hip/hip_bf16.h>
#include <math.h>
#include <stdint.h>

// Problem constants from the reference
#define NN   50000   // nodes
#define DIM  256     // feature dim (both in and out)
#define RELS 4       // relations
#define NEDG 500000  // edges per relation
#define NVUL 10000   // vulnerable nodes
#define KC   8       // candidates per node
#define EPSV 1e-8f

// Padded LDS row for the staged B stripe: 256 bf16 + 8 bf16 pad = 528 bytes.
// 528/4 = 132 dwords == 4 (mod 64 banks): 16 consecutive columns hit 16
// distinct 4-bank groups -> conflict-free ds_load_b128.
#define BROW 264

typedef __attribute__((ext_vector_type(16))) __bf16 v16bf;
typedef __attribute__((ext_vector_type(8)))  __bf16 v8bf;
typedef __attribute__((ext_vector_type(8)))  float  v8f;

// ---------------------------------------------------------------- prep kernels

__global__ __launch_bounds__(256) void k_cvt_x(const float* __restrict__ x,
                                               __bf16* __restrict__ xb, int n) {
  int i = blockIdx.x * 256 + threadIdx.x;
  if (i < n) xb[i] = (__bf16)x[i];
}

// Build transposed bf16 weights: wt[p][n][k] = W_p[k][n]; p=0..3 -> We[p], p=4 -> Wn.
__global__ __launch_bounds__(256) void k_build_wt(const float* __restrict__ We,
                                                  const float* __restrict__ Wn,
                                                  __bf16* __restrict__ wt) {
  int i = blockIdx.x * 256 + threadIdx.x;
  if (i >= 5 * DIM * DIM) return;
  int p   = i / (DIM * DIM);
  int rem = i % (DIM * DIM);
  int n   = rem / DIM;   // output column
  int k   = rem % DIM;   // input dim
  float v = (p < 4) ? We[(size_t)p * DIM * DIM + (size_t)k * DIM + n]
                    : Wn[(size_t)k * DIM + n];
  wt[i] = (__bf16)v;
}

__global__ __launch_bounds__(256) void k_zero(float* __restrict__ p, size_t n) {
  size_t i = (size_t)blockIdx.x * 256 + threadIdx.x;
  size_t stride = (size_t)gridDim.x * 256;
  for (; i < n; i += stride) p[i] = 0.f;
}

// ---------------------------------------------------------------- WMMA GEMM
// hall[p][n][d] = (x @ W_p)[n][d]  (+ bn[d] for p == 4)
// Block = 8 waves; each wave owns a 32-row x 64-col output tile (2 M-subtiles
// sharing B fragments) -> 256 rows x 64 cols per block.
// The 64x256 bf16 B stripe (32KB) is staged to LDS ONCE per block via
// gfx1250 async-to-LDS loads (ASYNCcnt), then fed to WMMAs through
// conflict-free ds_load_b128. Per K-step: 4 global b128 (A), 8 ds b128 (B),
// 8 v_wmma_f32_16x16x32_bf16.
__global__ __launch_bounds__(256) void k_gemm_wmma(
    const __bf16* __restrict__ xb,    // NN x DIM, row-major bf16
    const __bf16* __restrict__ wt,    // 5 x DIM(n) x DIM(k) bf16 (transposed)
    const float*  __restrict__ bn,    // DIM
    float*        __restrict__ hall)  // 5 x NN x DIM f32
{
  __shared__ __bf16 ldsB[64 * BROW];  // 64 cols x padded 256-K rows = 33 KB

  int t    = threadIdx.x;
  int lane = t & 31;
  int wave = t >> 5;
  int colBase = blockIdx.y * 64;
  int p = blockIdx.z;
  const __bf16* wp = wt + (size_t)p * DIM * DIM;

  // ---- cooperative async stage of the B stripe: 256 threads x 128B each ----
  {
    int srow = t >> 2;                 // 0..63 : column of the stripe
    int sseg = (t & 3) * 128;          // byte segment within the 512B row
    const char* gsrc = (const char*)(wp + (size_t)(colBase + srow) * DIM) + sseg;
    char* ldst = (char*)&ldsB[0] + srow * (BROW * 2) + sseg;
#pragma unroll
    for (int j = 0; j < 8; ++j) {
      uint32_t loff = (uint32_t)(uintptr_t)(ldst + j * 16);  // LDS byte offset
      const void* ga = gsrc + j * 16;
      asm volatile("global_load_async_to_lds_b128 %0, %1, off"
                   :: "v"(loff), "v"(ga) : "memory");
    }
    asm volatile("s_wait_asynccnt 0x0" ::: "memory");
  }
  __syncthreads();

  int rowBase = (blockIdx.x * 8 + wave) * 32;
  if (rowBase < NN) {                        // wave-uniform: EXEC stays all-ones
    int aSel = (lane >> 4) * 8;
    int aRow0 = rowBase + (lane & 15);
    int aRow1 = rowBase + 16 + (lane & 15);
    if (aRow0 >= NN) aRow0 = NN - 1;
    if (aRow1 >= NN) aRow1 = NN - 1;
    const __bf16* abase0 = xb + (size_t)aRow0 * DIM + aSel;
    const __bf16* abase1 = xb + (size_t)aRow1 * DIM + aSel;

    int bCol  = lane & 15;
    int bKsel = (lane >> 4) * 8;

    v8f acc[2][4] = {};

    for (int k0 = 0; k0 < DIM; k0 += 32) {
      // A frags (16x32 bf16): elems 0..7 = K[sel..sel+7], 8..15 = K[sel+16..+23]
      v8bf a0lo = *(const v8bf*)(abase0 + k0);
      v8bf a0hi = *(const v8bf*)(abase0 + k0 + 16);
      v8bf a1lo = *(const v8bf*)(abase1 + k0);
      v8bf a1hi = *(const v8bf*)(abase1 + k0 + 16);
      v16bf a0, a1;
#pragma unroll
      for (int i = 0; i < 8; ++i) {
        a0[i] = a0lo[i]; a0[i + 8] = a0hi[i];
        a1[i] = a1lo[i]; a1[i + 8] = a1hi[i];
      }

      // Preload ALL B frags for this K-step so ds loads batch into one clause.
      v16bf bfrag[4];
#pragma unroll
      for (int c = 0; c < 4; ++c) {
        const __bf16* bp = &ldsB[(c * 16 + bCol) * BROW + k0 + bKsel];
        v8bf blo = *(const v8bf*)(bp);
        v8bf bhi = *(const v8bf*)(bp + 16);
#pragma unroll
        for (int i = 0; i < 8; ++i) { bfrag[c][i] = blo[i]; bfrag[c][i + 8] = bhi[i]; }
      }

#pragma unroll
      for (int c = 0; c < 4; ++c)
        acc[0][c] = __builtin_amdgcn_wmma_f32_16x16x32_bf16(
            false, a0, false, bfrag[c], (short)0, acc[0][c], false, false);
#pragma unroll
      for (int c = 0; c < 4; ++c)
        acc[1][c] = __builtin_amdgcn_wmma_f32_16x16x32_bf16(
            false, a1, false, bfrag[c], (short)0, acc[1][c], false, false);
    }

    // C/D layout: VGPR v, lanes 0-15 -> row v, col lane; lanes 16-31 -> row v+8.
    float* outp = hall + (size_t)p * NN * DIM;
    int rowOff = (lane >> 4) * 8;
#pragma unroll
    for (int m = 0; m < 2; ++m) {
      int mBase = rowBase + m * 16 + rowOff;
#pragma unroll
      for (int c = 0; c < 4; ++c) {
        int col = colBase + c * 16 + bCol;
        float bias = (p == 4) ? bn[col] : 0.f;
#pragma unroll
        for (int v = 0; v < 8; ++v) {
          int row = mBase + v;
          if (row < NN) outp[(size_t)row * DIM + col] = acc[m][c][v] + bias;
        }
      }
    }
  }
}

// ---------------------------------------------------------------- edge phase

__device__ inline float waveSum(float v) {
#pragma unroll
  for (int off = 16; off > 0; off >>= 1) v += __shfl_xor(v, off, 32);
  return v;
}

// One wave32 per edge: cosine(hr[src], ht[dst]); scatter s*hsrc into hsum[dst].
__global__ __launch_bounds__(256) void k_edge(
    const float* __restrict__ hall,   // 5 x NN x DIM (p=relation, p=4 -> ht)
    const int*   __restrict__ src,    // RELS x NEDG
    const int*   __restrict__ dst,    // RELS x NEDG
    float*       __restrict__ hsum,   // RELS x NN x DIM (atomics)
    float*       __restrict__ asum,   // RELS x NN
    float*       __restrict__ cnt)    // RELS x NN
{
  int lane = threadIdx.x & 31;
  int wid  = threadIdx.x >> 5;
  int r = blockIdx.y;
  long e = (long)blockIdx.x * 8 + wid;
  if (e >= NEDG) return;

  int sn = src[(size_t)r * NEDG + e];
  int dn = dst[(size_t)r * NEDG + e];

  const float* hs = hall + ((size_t)r * NN + sn) * DIM;
  const float* hd = hall + ((size_t)4 * NN + dn) * DIM;

  float num = 0.f, nss = 0.f, ndd = 0.f;
  float vs[8];
#pragma unroll
  for (int i = 0; i < 8; ++i) {
    float a = hs[lane + i * 32];
    float b = hd[lane + i * 32];
    vs[i] = a;
    num += a * b; nss += a * a; ndd += b * b;
  }
  num = waveSum(num); nss = waveSum(nss); ndd = waveSum(ndd);
  float s = num / (fmaxf(sqrtf(nss), EPSV) * fmaxf(sqrtf(ndd), EPSV));

  float* out = hsum + ((size_t)r * NN + dn) * DIM;
#pragma unroll
  for (int i = 0; i < 8; ++i) atomicAdd(&out[lane + i * 32], s * vs[i]);
  if (lane == 0) {
    atomicAdd(&asum[(size_t)r * NN + dn], s);
    atomicAdd(&cnt [(size_t)r * NN + dn], 1.f);
  }
}

// ---------------------------------------------------------------- vuln mask
// One 256-thread block per vulnerable node. d2 tiles cached in 32KB LDS.
__global__ __launch_bounds__(256) void k_mask(
    const float* __restrict__ hsum,   // RELS x NN x DIM (read only here)
    const int*   __restrict__ cand,   // NVUL x KC
    float*       __restrict__ hmask)  // NVUL x RELS x DIM (masked Hv)
{
  int n = blockIdx.x;
  int t = threadIdx.x;
  int lane = t & 31, wid = t >> 5;

  __shared__ float d2s[KC * RELS * DIM];  // 8 * 1024 floats = 32 KB
  __shared__ float red[8];
  __shared__ float distS[KC];
  __shared__ float att[KC];

  float hv[4];
#pragma unroll
  for (int j = 0; j < 4; ++j) {            // j == relation r, t == d
    hv[j] = hsum[((size_t)j * NN + n) * DIM + t];
  }

  for (int k = 0; k < KC; ++k) {
    int c = cand[(size_t)n * KC + k];
    float part = 0.f;
#pragma unroll
    for (int j = 0; j < 4; ++j) {
      float diff = hv[j] - hsum[((size_t)j * NN + c) * DIM + t];
      float dd = diff * diff;
      d2s[k * 1024 + j * 256 + t] = dd;
      part += dd;
    }
    part = waveSum(part);
    if (lane == 0) red[wid] = part;
    __syncthreads();
    if (t == 0) {
      float s = 0.f;
      for (int w = 0; w < 8; ++w) s += red[w];
      distS[k] = sqrtf(s);
    }
    __syncthreads();
  }

  if (t == 0) {
    float m = -3.4e38f;
    for (int k = 0; k < KC; ++k) m = fmaxf(m, -distS[k]);
    float se = 0.f;
    for (int k = 0; k < KC; ++k) { float e = expf(-distS[k] - m); att[k] = e; se += e; }
    for (int k = 0; k < KC; ++k) att[k] /= se;
  }
  __syncthreads();

#pragma unroll
  for (int j = 0; j < 4; ++j) {
    float a = 0.f;
#pragma unroll
    for (int k = 0; k < KC; ++k) a += att[k] * d2s[k * 1024 + j * 256 + t];
    hmask[((size_t)n * RELS + j) * DIM + t] = hv[j] * expf(-a);
  }
}

// ---------------------------------------------------------------- combine
// out[n][d] = sum_r softmax_r(ma[n])[r] * h[n][r][d]
__global__ __launch_bounds__(256) void k_combine(
    const float* __restrict__ hsum,   // RELS x NN x DIM
    const float* __restrict__ hmask,  // NVUL x RELS x DIM
    const float* __restrict__ asum,   // RELS x NN
    const float* __restrict__ cnt,    // RELS x NN
    float*       __restrict__ out)    // NN x DIM
{
  int n = blockIdx.x;
  int d = threadIdx.x;
  float w[RELS];
  float m = -3.4e38f;
#pragma unroll
  for (int r = 0; r < RELS; ++r) {
    float ma = asum[(size_t)r * NN + n] / fmaxf(cnt[(size_t)r * NN + n], 1.f);
    w[r] = ma; m = fmaxf(m, ma);
  }
  float se = 0.f;
#pragma unroll
  for (int r = 0; r < RELS; ++r) { w[r] = expf(w[r] - m); se += w[r]; }
  float o = 0.f;
#pragma unroll
  for (int r = 0; r < RELS; ++r) {
    float hval = (n < NVUL) ? hmask[((size_t)n * RELS + r) * DIM + d]
                            : hsum[((size_t)r * NN + n) * DIM + d];
    o += (w[r] / se) * hval;
  }
  out[(size_t)n * DIM + d] = o;
}

// ---------------------------------------------------------------- launch

extern "C" void kernel_launch(void* const* d_in, const int* in_sizes, int n_in,
                              void* d_out, int out_size, void* d_ws, size_t ws_size,
                              hipStream_t stream) {
  const float* x    = (const float*)d_in[0];   // N x D
  const float* We   = (const float*)d_in[1];   // R x D x DO
  const float* Wn   = (const float*)d_in[2];   // D x DO
  const float* bn   = (const float*)d_in[3];   // DO
  const int*   src  = (const int*)  d_in[4];   // R x E
  const int*   dst  = (const int*)  d_in[5];   // R x E
  const int*   cand = (const int*)  d_in[6];   // NV x K
  // d_in[7] = n_vul scalar (known constant NVUL)

  char* ws = (char*)d_ws;
  size_t off = 0;
  auto carve = [&](size_t bytes) -> char* {
    char* p = ws + off;
    off = (off + bytes + 255) & ~(size_t)255;
    return p;
  };

  __bf16* xb    = (__bf16*)carve((size_t)NN * DIM * sizeof(__bf16));
  __bf16* wt    = (__bf16*)carve((size_t)5 * DIM * DIM * sizeof(__bf16));
  float*  hall  = (float*) carve((size_t)5 * NN * DIM * sizeof(float));
  float*  hsum  = (float*) carve((size_t)RELS * NN * DIM * sizeof(float));
  float*  asum  = (float*) carve((size_t)RELS * NN * sizeof(float));
  float*  cntb  = (float*) carve((size_t)RELS * NN * sizeof(float));
  float*  hmask = (float*) carve((size_t)NVUL * RELS * DIM * sizeof(float));

  // 1. precision conversion + weight transpose
  k_cvt_x<<<(NN * DIM + 255) / 256, 256, 0, stream>>>(x, xb, NN * DIM);
  k_build_wt<<<(5 * DIM * DIM + 255) / 256, 256, 0, stream>>>(We, Wn, wt);

  // 2. zero accumulators (must happen every call: graph is replayed)
  k_zero<<<2048, 256, 0, stream>>>(hsum, (size_t)RELS * NN * DIM);
  k_zero<<<256, 256, 0, stream>>>(asum, (size_t)RELS * NN);
  k_zero<<<256, 256, 0, stream>>>(cntb, (size_t)RELS * NN);

  // 3. all five projections via bf16 WMMA (B staged to LDS asynchronously)
  dim3 gg((NN + 255) / 256, DIM / 64, 5);
  k_gemm_wmma<<<gg, 256, 0, stream>>>(xb, wt, bn, hall);

  // 4. edge cosine attention + segment sums (wave per edge)
  dim3 ge((NEDG + 7) / 8, RELS);
  k_edge<<<ge, 256, 0, stream>>>(hall, src, dst, hsum, asum, cntb);

  // 5. vulnerable-node distance softmax + mask
  k_mask<<<NVUL, 256, 0, stream>>>(hsum, cand, hmask);

  // 6. relation softmax + weighted combine
  k_combine<<<NN, 256, 0, stream>>>(hsum, hmask, asum, cntb, (float*)d_out);
}
